// CombSub_16020228014836
// MI455X (gfx1250) — compile-verified, matrix operand order checked by reading, not run on hardware
//
#include <hip/hip_runtime.h>
#include <math.h>
#include <stdint.h>

// ---------------------------------------------------------------------------
// CombSub vocoder for MI455X (gfx1250, wave32).
//  - mel2ctrl GEMMs: v_wmma_f32_16x16x32_bf16, fragment-swizzled operands,
//    16x64 tile per wave (4 accumulators, contiguous 32B fragment loads).
//  - irfft's: f32 GEMM against precomputed DFT basis (v_wmma_f32_16x16x4_f32),
//    roll + normalization + fixed Hann baked into basis columns.
//  - fft_convolve: time-domain OLA scatter; IR row staged to LDS with
//    global_load_async_to_lds_b128 (ASYNCcnt) inline asm.
// ---------------------------------------------------------------------------

typedef __attribute__((ext_vector_type(16))) __bf16 v16bf;
typedef __attribute__((ext_vector_type(8)))  float  v8f;
typedef __attribute__((ext_vector_type(2)))  float  v2f;

#define NB      16
#define NFR     400
#define BLK     512
#define TLEN    (NFR * BLK)          // 204800
#define NMELS   80
#define HIDC    256
#define NAP     256
#define NHARM   512
#define NNOISE  256
#define NOUTC   1024
#define MROWS   (NB * NFR)           // 6400
#define PADLEN  206848               // >= 400*512 + 2045
#define PI_F    3.14159265358979323846f
#define PI_D    3.14159265358979323846

static inline int cdiv(long a, long b) { return (int)((a + b - 1) / b); }

// ---------------- phase / combtooth --------------------------------------

__global__ void k_frame_scan(const float* __restrict__ f0, double* __restrict__ Sph) {
  int b = blockIdx.x;
  if (threadIdx.x != 0) return;
  double s = 0.0;
  Sph[b * (NFR + 1)] = 0.0;
  for (int n = 0; n < NFR; ++n) {
    double f = (double)f0[b * NFR + n];
    double d = (n < NFR - 1) ? ((double)f0[b * NFR + n + 1] - f) : 0.0;
    s += 512.0 * f + 255.5 * d;
    Sph[b * (NFR + 1) + n + 1] = s;
  }
}

__global__ void k_phase_comb(const float* __restrict__ f0, const float* __restrict__ ph0,
                             const double* __restrict__ Sph,
                             float* __restrict__ comb, float* __restrict__ phaseF,
                             float* __restrict__ outPhase) {
  long i = (long)blockIdx.x * blockDim.x + threadIdx.x;
  if (i >= (long)NB * TLEN) return;
  int b = (int)(i / TLEN);
  int tt = (int)(i % TLEN);
  int n = tt / BLK, k = tt % BLK;
  float f = f0[b * NFR + n];
  float d = (n < NFR - 1) ? (f0[b * NFR + n + 1] - f) : 0.f;
  double partial = (double)(k + 1) * (double)f +
                   (double)d * ((double)k * (double)(k + 1) * 0.5) / 512.0;
  double x = (Sph[b * (NFR + 1) + n] + partial) / 44100.0 +
             (double)ph0[b] / (2.0 * PI_D);
  float xw = (float)(x - rint(x));
  float f0up = f + d * ((float)k / 512.f);
  float z = 44100.f * xw / (f0up + 0.001f);
  float s = PI_F * z;
  comb[i] = (fabsf(z) < 1e-12f) ? 1.f : (sinf(s) / s);
  if (k == 0) {
    float pf = 2.f * PI_F * xw;
    phaseF[b * NFR + n] = pf;
    outPhase[b * NFR + n] = pf;
  }
}

// ---------------- fragment swizzle helpers --------------------------------
// bf16 16x16x32: lane L holds row (L&15); its 16 elements cover K offsets
//   (e&7) + (L>=16 ? 8:0) + (e>=8 ? 16:0)  within a 32-wide K panel.
// f32 16x16x4: lane L holds row (L&15); 2 elements cover K offsets
//   (L>=16 ? 2:0) + e  within a 4-wide K panel.

__device__ __forceinline__ int bkoff(int e, int lane) {
  return (e & 7) + ((lane & 16) ? 8 : 0) + ((e & 8) ? 16 : 0);
}

// ---------------- bf16 packing (fragment-ordered im2col + weights) --------

// A layout: frag index ((tm*(K/32)+p)*32+lane), 16 bf16 each, row-contiguous.
__global__ void k_pack_a1(const float* __restrict__ mel, __bf16* __restrict__ A1) {
  long i = (long)blockIdx.x * blockDim.x + threadIdx.x;   // M*256 elements
  if (i >= (long)MROWS * 256) return;
  int e = (int)(i & 15), lane = (int)((i >> 4) & 31);
  int p = (int)((i >> 9) & 7), tm = (int)(i >> 12);       // K/32 = 8
  int m = (tm << 4) + (lane & 15);
  int k = (p << 5) + bkoff(e, lane);
  int b = m / NFR, l = m % NFR;
  float v = 0.f;
  if (k < 3 * NMELS) {
    int kk = k / NMELS, ci = k % NMELS;
    int lp = l + kk - 1;
    if (lp >= 0 && lp < NFR) v = mel[((long)b * NFR + lp) * NMELS + ci];
  }
  A1[i] = (__bf16)v;
}

__global__ void k_pack_a2(const float* __restrict__ h, __bf16* __restrict__ A2) {
  long i = (long)blockIdx.x * blockDim.x + threadIdx.x;   // M*768 elements
  if (i >= (long)MROWS * 768) return;
  int e = (int)(i & 15), lane = (int)((i >> 4) & 31);
  long rest = i >> 9;
  int p = (int)(rest % 24), tm = (int)(rest / 24);        // K/32 = 24
  int m = (tm << 4) + (lane & 15);
  int k = (p << 5) + bkoff(e, lane);
  int b = m / NFR, l = m % NFR;
  int kk = k / HIDC, c = k % HIDC;
  int lp = l + kk - 1;
  float v = (lp >= 0 && lp < NFR) ? h[((long)b * NFR + lp) * HIDC + c] : 0.f;
  A2[i] = (__bf16)v;
}

// B layout: frag index ((p*(N/16)+t16)*32+lane), 16 bf16 each.
__global__ void k_pack_w1(const float* __restrict__ w, __bf16* __restrict__ W1) {
  long i = (long)blockIdx.x * blockDim.x + threadIdx.x;   // 256*256
  if (i >= 256 * 256) return;
  int e = (int)(i & 15), lane = (int)((i >> 4) & 31);
  int t16 = (int)((i >> 9) & 15), p = (int)(i >> 13);
  int k = (p << 5) + bkoff(e, lane);
  int col = (t16 << 4) + (lane & 15);
  float v = 0.f;
  if (k < 3 * NMELS) {
    int kk = k / NMELS, ci = k % NMELS;
    v = w[(long)col * (NMELS * 3) + ci * 3 + kk];
  }
  W1[i] = (__bf16)v;
}

__global__ void k_pack_w2(const float* __restrict__ w, __bf16* __restrict__ W2) {
  long i = (long)blockIdx.x * blockDim.x + threadIdx.x;   // 768*256
  if (i >= 768 * 256) return;
  int e = (int)(i & 15), lane = (int)((i >> 4) & 31);
  long rest = i >> 9;
  int t16 = (int)(rest & 15), p = (int)(rest >> 4);       // p < 24
  int k = (p << 5) + bkoff(e, lane);
  int col = (t16 << 4) + (lane & 15);
  int kk = k / HIDC, c = k % HIDC;
  W2[i] = (__bf16)w[(long)col * 768 + c * 3 + kk];
}

__global__ void k_pack_w3(const float* __restrict__ w, __bf16* __restrict__ W3) {
  long i = (long)blockIdx.x * blockDim.x + threadIdx.x;   // 256*1024
  if (i >= 256 * 1024) return;
  int e = (int)(i & 15), lane = (int)((i >> 4) & 31);
  int t16 = (int)((i >> 9) & 63), p = (int)(i >> 15);
  int k = (p << 5) + bkoff(e, lane);
  int col = (t16 << 4) + (lane & 15);
  W3[i] = (__bf16)w[(long)col * 256 + k];
}

// ---------------- bf16 WMMA GEMM: 16x64 tile per wave ---------------------

__global__ void k_gemm_bf16(const __bf16* __restrict__ Ap, const __bf16* __restrict__ Bp,
                            const float* __restrict__ bias, float* __restrict__ C,
                            int M, int N, int K) {
  int wave = threadIdx.x >> 5, lane = threadIdx.x & 31;
  int tile = blockIdx.x * (blockDim.x >> 5) + wave;
  int tilesN = N >> 6;
  int tilesTot = (M >> 4) * tilesN;
  if (tile >= tilesTot) return;                 // uniform per wave
  int tm = tile / tilesN, tn = tile % tilesN;
  int panels = K >> 5, n16 = N >> 4;
  v8f acc0 = {}, acc1 = {}, acc2 = {}, acc3 = {};
  const v16bf* Af = (const v16bf*)Ap + (long)tm * panels * 32 + lane;
  for (int p = 0; p < panels; ++p) {
    v16bf a = Af[(long)p * 32];
    const v16bf* Bf = (const v16bf*)Bp + ((long)p * n16 + (tn << 2)) * 32 + lane;
    v16bf b0 = Bf[0], b1 = Bf[32], b2 = Bf[64], b3 = Bf[96];
    acc0 = __builtin_amdgcn_wmma_f32_16x16x32_bf16(false, a, false, b0, (short)0, acc0, false, false);
    acc1 = __builtin_amdgcn_wmma_f32_16x16x32_bf16(false, a, false, b1, (short)0, acc1, false, false);
    acc2 = __builtin_amdgcn_wmma_f32_16x16x32_bf16(false, a, false, b2, (short)0, acc2, false, false);
    acc3 = __builtin_amdgcn_wmma_f32_16x16x32_bf16(false, a, false, b3, (short)0, acc3, false, false);
  }
  int colb = (tn << 6) + (lane & 15);
  int rowb = (tm << 4) + ((lane < 16) ? 0 : 8);
#define ST_B(ACC, S) do { int col = colb + ((S) << 4); \
    float bv = bias ? bias[col] : 0.f; \
    _Pragma("unroll") for (int r = 0; r < 8; ++r) \
      C[(long)(rowb + r) * N + col] = ACC[r] + bv; } while (0)
  ST_B(acc0, 0); ST_B(acc1, 1); ST_B(acc2, 2); ST_B(acc3, 3);
#undef ST_B
}

// ---------------- f32 WMMA GEMM (DFT): 16x64 tile per wave ----------------
// A row-major f32 [M,K]; B pre-packed fragment order: ((p*(N/16)+t16)*32+lane)*2.

__global__ void k_gemm_f32(const float* __restrict__ A, const float* __restrict__ Bp,
                           float* __restrict__ C, int M, int N, int K) {
  int wave = threadIdx.x >> 5, lane = threadIdx.x & 31;
  int tile = blockIdx.x * (blockDim.x >> 5) + wave;
  int tilesN = N >> 6;
  int tilesTot = (M >> 4) * tilesN;
  if (tile >= tilesTot) return;
  int tm = tile / tilesN, tn = tile % tilesN;
  int row = (tm << 4) + (lane & 15);
  int koff = (lane & 16) ? 2 : 0;
  int panels = K >> 2, n16 = N >> 4;
  v8f acc0 = {}, acc1 = {}, acc2 = {}, acc3 = {};
  for (int p = 0; p < panels; ++p) {
    v2f a = *(const v2f*)&A[(long)row * K + (p << 2) + koff];
    const v2f* Bf = (const v2f*)Bp + ((long)p * n16 + (tn << 2)) * 32 + lane;
    v2f b0 = Bf[0], b1 = Bf[32], b2 = Bf[64], b3 = Bf[96];
    acc0 = __builtin_amdgcn_wmma_f32_16x16x4_f32(false, a, false, b0, (short)0, acc0, false, false);
    acc1 = __builtin_amdgcn_wmma_f32_16x16x4_f32(false, a, false, b1, (short)0, acc1, false, false);
    acc2 = __builtin_amdgcn_wmma_f32_16x16x4_f32(false, a, false, b2, (short)0, acc2, false, false);
    acc3 = __builtin_amdgcn_wmma_f32_16x16x4_f32(false, a, false, b3, (short)0, acc3, false, false);
  }
  int colb = (tn << 6) + (lane & 15);
  int rowb = (tm << 4) + ((lane < 16) ? 0 : 8);
#define ST_F(ACC, S) do { int col = colb + ((S) << 4); \
    _Pragma("unroll") for (int r = 0; r < 8; ++r) \
      C[(long)(rowb + r) * N + col] = ACC[r]; } while (0)
  ST_F(acc0, 0); ST_F(acc1, 1); ST_F(acc2, 2); ST_F(acc3, 3);
#undef ST_F
}

// ---------------- DFT basis generators (packed f32 fragment order) --------
// decode i -> (p, t16, lane, e); k = p*4 + (lane&16?2:0) + e; col = t16*16+(lane&15)

// allpass basis: K=512 ([Re 0..255 | Im 0..255]), N=512 (510 used)
__global__ void k_basis_allpass(float* __restrict__ B) {
  long i = (long)blockIdx.x * blockDim.x + threadIdx.x;
  if (i >= 512L * 512) return;
  int e = (int)(i & 1), lane = (int)((i >> 1) & 31);
  int t16 = (int)((i >> 6) & 31), p = (int)(i >> 11);
  int k = (p << 2) + ((lane & 16) ? 2 : 0) + e;
  int col = (t16 << 4) + (lane & 15);
  float v = 0.f;
  if (col < 510) {
    int tp = (col + 255) % 510;                     // roll(255)
    if (k < 256) {                                  // real part rows
      if (k == 0)        v = 1.f / 510.f;
      else if (k == 255) v = ((tp & 1) ? -1.f : 1.f) / 510.f;
      else {
        int j = (k * tp) % 510;
        v = (float)(2.0 * cos(2.0 * PI_D * (double)j / 510.0) / 510.0);
      }
    } else {                                        // imag part rows
      int kk = k - 256;
      if (kk >= 1 && kk <= 254) {
        int j = (kk * tp) % 510;
        v = (float)(-2.0 * sin(2.0 * PI_D * (double)j / 510.0) / 510.0);
      }
    }
  }
  B[i] = v;
}

// src basis: K=512 real bins, N=1024 (1022 used); window applied later
__global__ void k_basis_src(float* __restrict__ B) {
  long i = (long)blockIdx.x * blockDim.x + threadIdx.x;
  if (i >= 512L * 1024) return;
  int e = (int)(i & 1), lane = (int)((i >> 1) & 31);
  int t16 = (int)((i >> 6) & 63), p = (int)(i >> 12);
  int k = (p << 2) + ((lane & 16) ? 2 : 0) + e;
  int col = (t16 << 4) + (lane & 15);
  float v = 0.f;
  if (col < 1022) {
    int tp = (col + 511) % 1022;                    // roll(511)
    if (k == 0)        v = 1.f / 1022.f;
    else if (k == 511) v = ((tp & 1) ? -1.f : 1.f) / 1022.f;
    else {
      int j = (k * tp) % 1022;
      v = (float)(2.0 * cos(2.0 * PI_D * (double)j / 1022.0) / 1022.0);
    }
  }
  B[i] = v;
}

// noise basis: K=256 real bins, N=512 (510 used); periodic hann baked in
__global__ void k_basis_noise(float* __restrict__ B) {
  long i = (long)blockIdx.x * blockDim.x + threadIdx.x;
  if (i >= 256L * 512) return;
  int e = (int)(i & 1), lane = (int)((i >> 1) & 31);
  int t16 = (int)((i >> 6) & 31), p = (int)(i >> 11);
  int k = (p << 2) + ((lane & 16) ? 2 : 0) + e;
  int col = (t16 << 4) + (lane & 15);
  float v = 0.f;
  if (col < 510) {
    int tp = (col + 255) % 510;
    double w = 0.5 - 0.5 * cos(2.0 * PI_D * (double)col / 510.0);  // hann[col]
    double raw;
    if (k == 0)        raw = 1.0 / 510.0;
    else if (k == 255) raw = ((tp & 1) ? -1.0 : 1.0) / 510.0;
    else {
      int j = (k * tp) % 510;
      raw = 2.0 * cos(2.0 * PI_D * (double)j / 510.0) / 510.0;
    }
    v = (float)(raw * w);
  }
  B[i] = v;
}

// ---------------- GroupNorm(4) + leaky ReLU (in place) --------------------

__global__ void k_groupnorm(float* __restrict__ h, const float* __restrict__ gg,
                            const float* __restrict__ gb) {
  int b = blockIdx.x >> 2, g = blockIdx.x & 3;
  __shared__ float r1[256], r2[256];
  float s = 0.f, s2 = 0.f;
  for (int idx = threadIdx.x; idx < NFR * 64; idx += blockDim.x) {
    int l = idx >> 6, c = (g << 6) + (idx & 63);
    float v = h[((long)b * NFR + l) * HIDC + c];
    s += v; s2 += v * v;
  }
  r1[threadIdx.x] = s; r2[threadIdx.x] = s2;
  __syncthreads();
  for (int off = 128; off > 0; off >>= 1) {
    if ((int)threadIdx.x < off) { r1[threadIdx.x] += r1[threadIdx.x + off];
                                  r2[threadIdx.x] += r2[threadIdx.x + off]; }
    __syncthreads();
  }
  float mu  = r1[0] / (NFR * 64.f);
  float var = r2[0] / (NFR * 64.f) - mu * mu;
  float inv = rsqrtf(var + 1e-5f);
  __syncthreads();
  for (int idx = threadIdx.x; idx < NFR * 64; idx += blockDim.x) {
    int l = idx >> 6, c = (g << 6) + (idx & 63);
    long pidx = ((long)b * NFR + l) * HIDC + c;
    float v = (h[pidx] - mu) * inv * gg[c] + gb[c];
    h[pidx] = (v >= 0.f) ? v : 0.01f * v;
  }
}

// ---------------- phase embedding + LayerNorm -> swizzled bf16 A3 ---------

__global__ void k_pe_ln(const float* __restrict__ h2, const float* __restrict__ phaseF,
                        const float* __restrict__ pw, const float* __restrict__ pb,
                        const float* __restrict__ lg, const float* __restrict__ lb,
                        __bf16* __restrict__ A3) {
  int m = blockIdx.x, c = threadIdx.x;
  float phi = phaseF[m];
  float v = h2[(long)m * HIDC + c] + sinf(phi) * pw[c * 2] + cosf(phi) * pw[c * 2 + 1] + pb[c];
  __shared__ float red[HIDC];
  red[c] = v; __syncthreads();
  for (int off = 128; off > 0; off >>= 1) { if (c < off) red[c] += red[c + off]; __syncthreads(); }
  float mu = red[0] / HIDC; __syncthreads();
  float dv = v - mu;
  red[c] = dv * dv; __syncthreads();
  for (int off = 128; off > 0; off >>= 1) { if (c < off) red[c] += red[c + off]; __syncthreads(); }
  float var = red[0] / HIDC;
  float o = dv * rsqrtf(var + 1e-5f) * lg[c] + lb[c];
  // inverse fragment swizzle (K=256 -> 8 panels)
  int p = c >> 5, r = c & 31;
  int e = (r & 7) | ((r & 16) ? 8 : 0);
  int half = (r >> 3) & 1;
  long idx = ((((long)(m >> 4) * 8 + p) * 32) + (half << 4) + (m & 15)) * 16 + e;
  A3[idx] = (__bf16)o;
}

// -------- group-delay -> cumsum -> complex exp, packed [Re|Im] rows -------

__global__ void k_gd(const float* __restrict__ ctrl, float* __restrict__ Aap) {
  int m = blockIdx.x, c = threadIdx.x;
  float v = PI_F * tanhf(ctrl[(long)m * NOUTC + c]);
  __shared__ float sc[NAP];
  sc[c] = v; __syncthreads();
  for (int off = 1; off < NAP; off <<= 1) {
    float t = (c >= off) ? sc[c - off] : 0.f;
    __syncthreads();
    sc[c] += t;
    __syncthreads();
  }
  float s = sc[c];
  Aap[(long)m * 512 + c]       = cosf(s);
  Aap[(long)m * 512 + 256 + c] = sinf(s);
}

// ---------------- spectrum matrices + src window --------------------------

__global__ void k_exp_src(const float* __restrict__ ctrl, float* __restrict__ As) {
  long i = (long)blockIdx.x * blockDim.x + threadIdx.x;
  if (i >= (long)MROWS * 512) return;
  int m = (int)(i >> 9), k = (int)(i & 511);
  As[i] = expf(ctrl[(long)m * NOUTC + NAP + k]);
}

__global__ void k_exp_noise(const float* __restrict__ ctrl, float* __restrict__ An) {
  long i = (long)blockIdx.x * blockDim.x + threadIdx.x;
  if (i >= (long)MROWS * 256) return;
  int m = (int)(i >> 8), k = (int)(i & 255);
  An[i] = expf(ctrl[(long)m * NOUTC + NAP + NHARM + k]) * (1.f / 128.f);
}

__global__ void k_src_window(float* __restrict__ ir, const float* __restrict__ f0fr) {
  long i = (long)blockIdx.x * blockDim.x + threadIdx.x;
  if (i >= (long)MROWS * 1024) return;
  int m = (int)(i >> 10), c = (int)(i & 1023);
  if (c >= 1022) { ir[i] = 0.f; return; }
  float hw = 1.5f * 44100.f / (f0fr[m] + 0.001f);
  float wv = ((float)c - 511.f) / hw;
  float w = (wv > 1.f) ? 0.f : 0.5f * (1.f + cosf(PI_F * wv));
  ir[i] *= w;
}

// ---------------- OLA convolution scatter ---------------------------------

__global__ void k_conv_ola(const float* __restrict__ audio, const float* __restrict__ ir,
                           int irSize, int irStride, float* __restrict__ pad,
                           float scale, float shift) {
  int blk = blockIdx.x;
  int b = blk / (NFR + 1);
  int n = blk % (NFR + 1);
  __shared__ float fr[1024]; __shared__ float irl[1024];
  int irRow = b * NFR + min(n, NFR - 1);
  // IR row: async DMA global -> LDS (16B per lane-chunk), then wait ASYNCcnt.
  {
    unsigned ldsBase = (unsigned)(unsigned long long)(void*)irl;  // LDS byte offset
    const float* g = ir + (long)irRow * irStride;
    int nChunks = irStride >> 2;                 // irStride floats = stride/4 16B chunks
    for (int c = threadIdx.x; c < nChunks; c += blockDim.x) {
      unsigned lo = ldsBase + (unsigned)c * 16u;
      unsigned long long ga = (unsigned long long)(g + c * 4);
      asm volatile("global_load_async_to_lds_b128 %0, %1, off"
                   :: "v"(lo), "v"(ga) : "memory");
    }
    asm volatile("s_wait_asynccnt 0x0" ::: "memory");
  }
  for (int k = threadIdx.x; k < 1024; k += blockDim.x) {
    long s = (long)(n - 1) * BLK + k;
    float v = (s >= 0 && s < TLEN) ? audio[(long)b * TLEN + s] * scale + shift : 0.f;
    float bart = 1.f - fabsf((float)k / 512.f - 1.f);
    fr[k] = v * bart;
  }
  __syncthreads();
  int fft = 1024 + irSize - 1;
  for (int j = threadIdx.x; j < fft; j += blockDim.x) {
    int k0 = max(0, j - irSize + 1), k1 = min(1023, j);
    float acc = 0.f;
    for (int k = k0; k <= k1; ++k) acc += fr[k] * irl[j - k];
    atomicAdd(&pad[(long)b * PADLEN + (long)n * BLK + j], acc);
  }
}

__global__ void k_extract(const float* __restrict__ pad, float* __restrict__ out) {
  long i = (long)blockIdx.x * blockDim.x + threadIdx.x;
  if (i >= (long)NB * TLEN) return;
  int b = (int)(i / TLEN), t = (int)(i % TLEN);
  out[i] = pad[(long)b * PADLEN + BLK + t];
}

__global__ void k_add(const float* __restrict__ a, const float* __restrict__ b,
                      float* __restrict__ o) {
  long i = (long)blockIdx.x * blockDim.x + threadIdx.x;
  if (i >= (long)NB * TLEN) return;
  o[i] = a[i] + b[i];
}

// ---------------------------------------------------------------------------

extern "C" void kernel_launch(void* const* d_in, const int* in_sizes, int n_in,
                              void* d_out, int out_size, void* d_ws, size_t ws_size,
                              hipStream_t stream) {
  (void)in_sizes; (void)n_in; (void)out_size; (void)ws_size;
  const float* mel  = (const float*)d_in[0];
  const float* f0f  = (const float*)d_in[1];
  const float* ph0  = (const float*)d_in[2];
  const float* unoi = (const float*)d_in[3];
  const float* c1w  = (const float*)d_in[4];
  const float* c1b  = (const float*)d_in[5];
  const float* gg   = (const float*)d_in[6];
  const float* gb   = (const float*)d_in[7];
  const float* c2w  = (const float*)d_in[8];
  const float* c2b  = (const float*)d_in[9];
  const float* pw   = (const float*)d_in[10];
  const float* pb   = (const float*)d_in[11];
  const float* lg   = (const float*)d_in[12];
  const float* lb   = (const float*)d_in[13];
  const float* ow   = (const float*)d_in[14];
  const float* ob   = (const float*)d_in[15];

  float* outSig   = (float*)d_out;
  float* outPhase = outSig + (size_t)NB * TLEN;
  float* outHarm  = outPhase + (size_t)NB * NFR;
  float* outNoise = outHarm + (size_t)NB * TLEN;

  char* wsp = (char*)d_ws;
  auto take = [&](size_t bytes) -> char* {
    char* p = wsp; wsp += (bytes + 255) & ~(size_t)255; return p;
  };
  double* Sph    = (double*)take((size_t)NB * (NFR + 1) * sizeof(double));
  float*  phaseF = (float*) take((size_t)NB * NFR * 4);
  float*  comb   = (float*) take((size_t)NB * TLEN * 4);
  __bf16* A1     = (__bf16*)take((size_t)MROWS * 256 * 2);
  __bf16* W1     = (__bf16*)take((size_t)256 * 256 * 2);
  __bf16* A2     = (__bf16*)take((size_t)MROWS * 768 * 2);
  __bf16* W2     = (__bf16*)take((size_t)768 * 256 * 2);
  __bf16* A3     = (__bf16*)take((size_t)MROWS * 256 * 2);
  __bf16* W3     = (__bf16*)take((size_t)256 * 1024 * 2);
  float*  hbuf   = (float*) take((size_t)MROWS * HIDC * 4);
  float*  ctrl   = (float*) take((size_t)MROWS * NOUTC * 4);
  float*  bap    = (float*) take((size_t)512 * 512 * 4);
  float*  bsc    = (float*) take((size_t)512 * 1024 * 4);
  float*  bnz    = (float*) take((size_t)256 * 512 * 4);
  float*  irb    = (float*) take((size_t)MROWS * 1024 * 4);
  float*  harm1  = (float*) take((size_t)NB * TLEN * 4);
  float*  padb   = (float*) take((size_t)NB * PADLEN * 4);
  // spectrum matrices alias the OLA pad buffer (disjoint lifetimes)
  float*  Aap = padb;   // MROWS*512 f32 = 13.1MB <= 13.2MB
  float*  As  = padb;   // MROWS*512
  float*  An  = padb;   // MROWS*256

  const int TB = 256;
  long nsamp = (long)NB * TLEN;

  // 1) phase integral + combtooth + phase_frames
  k_frame_scan<<<NB, 32, 0, stream>>>(f0f, Sph);
  k_phase_comb<<<cdiv(nsamp, TB), TB, 0, stream>>>(f0f, ph0, Sph, comb, phaseF, outPhase);

  // DFT bases (packed fragment order)
  k_basis_allpass<<<cdiv(512L * 512, TB), TB, 0, stream>>>(bap);
  k_basis_src<<<cdiv(512L * 1024, TB), TB, 0, stream>>>(bsc);
  k_basis_noise<<<cdiv(256L * 512, TB), TB, 0, stream>>>(bnz);

  // 2) mel2ctrl (bf16 WMMA GEMMs, fragment-swizzled operands)
  k_pack_a1<<<cdiv((long)MROWS * 256, TB), TB, 0, stream>>>(mel, A1);
  k_pack_w1<<<cdiv(256L * 256, TB), TB, 0, stream>>>(c1w, W1);
  k_gemm_bf16<<<cdiv((MROWS / 16) * (256 / 64), 8), TB, 0, stream>>>(A1, W1, c1b, hbuf, MROWS, 256, 256);
  k_groupnorm<<<NB * 4, TB, 0, stream>>>(hbuf, gg, gb);
  k_pack_a2<<<cdiv((long)MROWS * 768, TB), TB, 0, stream>>>(hbuf, A2);
  k_pack_w2<<<cdiv(768L * 256, TB), TB, 0, stream>>>(c2w, W2);
  k_gemm_bf16<<<cdiv((MROWS / 16) * (256 / 64), 8), TB, 0, stream>>>(A2, W2, c2b, hbuf, MROWS, 256, 768);
  k_pe_ln<<<MROWS, HIDC, 0, stream>>>(hbuf, phaseF, pw, pb, lg, lb, A3);
  k_pack_w3<<<cdiv(256L * 1024, TB), TB, 0, stream>>>(ow, W3);
  k_gemm_bf16<<<cdiv((MROWS / 16) * (1024 / 64), 8), TB, 0, stream>>>(A3, W3, ob, ctrl, MROWS, 1024, 256);

  // 3) allpass IR (f32 WMMA DFT) -> conv with combtooth
  k_gd<<<MROWS, NAP, 0, stream>>>(ctrl, Aap);
  k_gemm_f32<<<cdiv((MROWS / 16) * (512 / 64), 8), TB, 0, stream>>>(Aap, bap, irb, MROWS, 512, 512);
  hipMemsetAsync(padb, 0, (size_t)NB * PADLEN * 4, stream);
  k_conv_ola<<<NB * (NFR + 1), TB, 0, stream>>>(comb, irb, 510, 512, padb, 1.f, 0.f);
  k_extract<<<cdiv(nsamp, TB), TB, 0, stream>>>(padb, harm1);

  // 4) source IR (f32 WMMA DFT + per-row window) -> conv with harmonic
  k_exp_src<<<cdiv((long)MROWS * 512, TB), TB, 0, stream>>>(ctrl, As);
  k_gemm_f32<<<cdiv((MROWS / 16) * (1024 / 64), 8), TB, 0, stream>>>(As, bsc, irb, MROWS, 1024, 512);
  k_src_window<<<cdiv((long)MROWS * 1024, TB), TB, 0, stream>>>(irb, f0f);
  hipMemsetAsync(padb, 0, (size_t)NB * PADLEN * 4, stream);
  k_conv_ola<<<NB * (NFR + 1), TB, 0, stream>>>(harm1, irb, 1022, 1024, padb, 1.f, 0.f);
  k_extract<<<cdiv(nsamp, TB), TB, 0, stream>>>(padb, outHarm);

  // 5) noise IR (f32 WMMA DFT, hann baked in) -> conv with noise
  k_exp_noise<<<cdiv((long)MROWS * 256, TB), TB, 0, stream>>>(ctrl, An);
  k_gemm_f32<<<cdiv((MROWS / 16) * (512 / 64), 8), TB, 0, stream>>>(An, bnz, irb, MROWS, 512, 256);
  hipMemsetAsync(padb, 0, (size_t)NB * PADLEN * 4, stream);
  k_conv_ola<<<NB * (NFR + 1), TB, 0, stream>>>(unoi, irb, 510, 512, padb, 2.f, -1.f);
  k_extract<<<cdiv(nsamp, TB), TB, 0, stream>>>(padb, outNoise);

  // 6) signal = harmonic + noise
  k_add<<<cdiv(nsamp, TB), TB, 0, stream>>>(outHarm, outNoise, outSig);
}
